// CorrNeigh_28424093565221
// MI455X (gfx1250) — matrix operand order repaired and checked by reading, not compiled, not run on hardware
//
#include <hip/hip_runtime.h>

typedef _Float16 h16 __attribute__((ext_vector_type(16)));
typedef _Float16 h8  __attribute__((ext_vector_type(8)));
typedef _Float16 h2  __attribute__((ext_vector_type(2)));
typedef float    f8  __attribute__((ext_vector_type(8)));

#define NIMG   8
#define CCH    256
#define HH     128
#define WW     128
#define HWSZ   (HH*WW)
#define PAD    3
#define HS     8            // h rows per block (one per wave)
#define WSTEP  10           // useful pixels per 16-wide tile
#define NWG    13           // ceil(128/10) w-groups
#define NHS    (HH/HS)      // 16 h-strips
#define CHUNK  64           // channels staged per LDS pass
#define CP     72           // padded channel pitch in halves (16B-aligned, bank-spread)
#define XROWS  HS           // 8 x rows
#define YROWS  (HS+6)       // 14 y rows (halo of +/-3)

__global__ __launch_bounds__(256)
void corr_volume_wmma(const float* __restrict__ x,
                      const float* __restrict__ y,
                      float* __restrict__ out) {
  __shared__ alignas(16) _Float16 Xs[XROWS * 16 * CP];
  __shared__ alignas(16) _Float16 Ys[YROWS * 16 * CP];

  const int bid  = blockIdx.x;
  const int wg   = bid % NWG;
  const int hsid = (bid / NWG) % NHS;
  const int nimg = bid / (NWG * NHS);
  const int w0 = wg * WSTEP;
  const int h0 = hsid * HS;

  const int tid  = threadIdx.x;
  const int lane = tid & 31;
  const int wv   = tid >> 5;        // wave id 0..7  -> owns h = h0+wv
  const int l    = tid & 15;        // column lane   (matrix N / M index)
  const int q    = (tid >> 4) & 15; // load-phase task slot
  const int g    = lane >> 4;       // half-wave group

  const size_t nbase = (size_t)nimg * CCH * HWSZ;

  f8 acc[7];
#pragma unroll
  for (int i = 0; i < 7; ++i) acc[i] = (f8)0.0f;

  for (int ch = 0; ch < CCH / CHUNK; ++ch) {
    const int c0 = ch * CHUNK;
    __syncthreads();   // WAR: previous chunk's readers done before overwrite

    // ---------------- stage f32 -> f16 into LDS (channel-major per column) --
    // 22 (row) x 32 (c-pair) tasks, 16 q-slots, 16 columns per task.
    for (int it = 0; it < (XROWS + YROWS) * (CHUNK / 2) / 16; ++it) {
      const int tsk = q + (it << 4);
      const int row = tsk >> 5;          // 0..21
      const int cp2 = tsk & 31;          // c-pair within chunk
      const int c   = c0 + (cp2 << 1);
      if (row < XROWS) {
        const int hh = h0 + row;
        int ww = w0 + l; ww = (ww > WW - 1) ? (WW - 1) : ww;   // clamp: rows m>=10 unused
        const float* p = x + nbase + (size_t)c * HWSZ + hh * WW + ww;
        h2 v; v.x = (_Float16)p[0]; v.y = (_Float16)p[HWSZ];
        *(h2*)&Xs[(row * 16 + l) * CP + (cp2 << 1)] = v;
      } else {
        const int r  = row - XROWS;           // 0..13
        const int hy = h0 + r - PAD;
        const int wy = w0 - PAD + l;
        const bool ok = (hy >= 0) && (hy < HH) && (wy >= 0) && (wy < WW);
        const int hc = hy < 0 ? 0 : (hy > HH - 1 ? HH - 1 : hy);
        const int wc = wy < 0 ? 0 : (wy > WW - 1 ? WW - 1 : wy);
        const float* p = y + nbase + (size_t)c * HWSZ + hc * WW + wc;
        h2 v;
        v.x = ok ? (_Float16)p[0]    : (_Float16)0.0f;   // zero-padded halo
        v.y = ok ? (_Float16)p[HWSZ] : (_Float16)0.0f;
        *(h2*)&Ys[(r * 16 + l) * CP + (cp2 << 1)] = v;
      }
    }
    __syncthreads();

    // ---------------- WMMA accumulate: D = Xs_tile^T * Ys_tile -------------
    const _Float16* xr = &Xs[(wv * 16 + l) * CP];
#pragma unroll
    for (int kk = 0; kk < 2; ++kk) {
      // A (16x32 f16): halves[0..7] = K 8g+0..7, halves[8..15] = K 16+8g+0..7
      h8 alo = *(const h8*)(xr + kk * 32 + 8 * g);
      h8 ahi = *(const h8*)(xr + kk * 32 + 16 + 8 * g);
      h16 a = __builtin_shufflevector(alo, ahi,
              0,1,2,3,4,5,6,7,8,9,10,11,12,13,14,15);

      // Preload ALL 7 B fragments; the sched_barrier below forbids the
      // scheduler from sinking these loads back between the WMMAs, so the
      // whole group issues, waits once on DScnt, then the 7 WMMAs run
      // back-to-back on the matrix pipe.
      h16 b[7];
#pragma unroll
      for (int i = 0; i < 7; ++i) {
        // B (32x16 f16): halves[0..7] = K 8g+0..7, halves[8..15] = K 16+8g+0..7 at column l
        const _Float16* yr = &Ys[((wv + i) * 16 + l) * CP + kk * 32 + 8 * g];
        h8 blo = *(const h8*)(yr);
        h8 bhi = *(const h8*)(yr + 16);
        b[i] = __builtin_shufflevector(blo, bhi,
               0,1,2,3,4,5,6,7,8,9,10,11,12,13,14,15);
      }
      __builtin_amdgcn_sched_barrier(0);   // loads above, WMMAs below
#pragma unroll
      for (int i = 0; i < 7; ++i) {
        acc[i] = __builtin_amdgcn_wmma_f32_16x16x32_f16(
                     false, a, false, b[i], (short)0, acc[i], false, false);
      }
      __builtin_amdgcn_sched_barrier(0);
    }
  }

  // ---------------- scatter the 7x7 band of each D tile ---------------------
  // D[m][n]: VGPR r, lane (g,l): m = r + 8g, n = l.  pixel w = w0+m, j = n-m.
  // Non-temporal: keep the 25.7MB output from evicting x/y in L2.
  const int h = h0 + wv;
#pragma unroll
  for (int i = 0; i < 7; ++i) {
#pragma unroll
    for (int r = 0; r < 8; ++r) {
      const int m = r + (g << 3);
      const int j = l - m;
      const int w = w0 + m;
      if (j >= 0 && j <= 6 && m < WSTEP && w < WW) {
        const size_t o = ((size_t)(nimg * 49 + i * 7 + j) * HH + h) * WW + w;
        __builtin_nontemporal_store(acc[i][r], &out[o]);
      }
    }
  }
}

extern "C" void kernel_launch(void* const* d_in, const int* in_sizes, int n_in,
                              void* d_out, int out_size, void* d_ws, size_t ws_size,
                              hipStream_t stream) {
  (void)in_sizes; (void)n_in; (void)d_ws; (void)ws_size; (void)out_size;
  const float* x = (const float*)d_in[0];
  const float* y = (const float*)d_in[1];
  float* out = (float*)d_out;

  dim3 grid(NIMG * NHS * NWG);   // 8 * 16 * 13 = 1664 blocks
  dim3 block(256);               // 8 wave32s
  corr_volume_wmma<<<grid, block, 0, stream>>>(x, y, out);
}
